// Controller_5085241279039
// MI455X (gfx1250) — matrix operand order repaired and checked by reading, not compiled, not run on hardware
//
#include <hip/hip_runtime.h>
#include <hip/hip_bf16.h>

#define B_TOT   4096
#define EMB     128
#define HID     512
#define OUTW    64
#define STEPS   20
#define GATES   (4 * HID)
#define ROWS    16
#define THREADS 512

typedef __attribute__((ext_vector_type(16))) __bf16 v16bf;
typedef __attribute__((ext_vector_type(8)))  float  v8f;

__device__ __forceinline__ unsigned short f2bf(float x) {
    unsigned int u = __float_as_uint(x);
    unsigned int r = ((u >> 16) & 1u) + 0x7FFFu;   // round-to-nearest-even
    return (unsigned short)((u + r) >> 16);
}

__device__ __forceinline__ float sigf(float x) { return 1.0f / (1.0f + __expf(-x)); }
__device__ __forceinline__ float tanhfast(float x) {
    float e = __expf(2.0f * x);
    return (e - 1.0f) / (e + 1.0f);
}

// fp32 -> bf16 conversion for W_ih, W_hh, embedding, W_lin (once per launch)
__global__ void convert_bf16_kernel(const float* __restrict__ wih,
                                    const float* __restrict__ whh,
                                    const float* __restrict__ emb,
                                    const float* __restrict__ wlin,
                                    unsigned short* __restrict__ owih,
                                    unsigned short* __restrict__ owhh,
                                    unsigned short* __restrict__ oemb,
                                    unsigned short* __restrict__ owlin) {
    const int n1 = GATES * EMB;
    const int n2 = GATES * HID;
    const int n3 = OUTW * EMB;
    const int n4 = OUTW * HID;
    int i = blockIdx.x * blockDim.x + threadIdx.x;
    if (i < n1)                     owih[i]                 = f2bf(wih[i]);
    else if (i < n1 + n2)           owhh[i - n1]            = f2bf(whh[i - n1]);
    else if (i < n1 + n2 + n3)      oemb[i - n1 - n2]       = f2bf(emb[i - n1 - n2]);
    else if (i < n1 + n2 + n3 + n4) owlin[i - n1 - n2 - n3] = f2bf(wlin[i - n1 - n2 - n3]);
}

// Build a 16-element bf16 fragment from two 16-byte chunks.
__device__ __forceinline__ v16bf load_frag16(const unsigned short* p0,
                                             const unsigned short* p1) {
    v16bf v;
    ((uint4*)&v)[0] = *(const uint4*)p0;
    ((uint4*)&v)[1] = *(const uint4*)p1;
    return v;
}

// One block = 16 batch rows for all 20 autoregressive steps.
// 16 waves; wave w owns hidden columns [32w, 32w+32) of every gate.
__global__ __launch_bounds__(THREADS)
void controller_kernel(const int* __restrict__ x,
                       const unsigned short* __restrict__ emb_bf,
                       const unsigned short* __restrict__ wih_bf,
                       const unsigned short* __restrict__ whh_bf,
                       const unsigned short* __restrict__ wlin_bf,
                       const float* __restrict__ b_ih,
                       const float* __restrict__ b_hh,
                       const float* __restrict__ b_lin,
                       float* __restrict__ out) {
    __shared__ __attribute__((aligned(16))) unsigned short h_bf_s[ROWS][HID];   // 16 KB
    __shared__ __attribute__((aligned(16))) float partial_s[4][ROWS][OUTW];     // 16 KB
    __shared__ __attribute__((aligned(16))) float logits_s[ROWS][OUTW];         // 4 KB
    __shared__ float blin_s[OUTW];
    __shared__ int   tok_s[ROWS];
    __shared__ float rmax_s[ROWS];
    __shared__ float rinv_s[ROWS];

    const int tid     = threadIdx.x;
    const int wave    = tid >> 5;
    const int lane    = tid & 31;
    const int ln      = lane & 15;
    const int laneHi  = lane >> 4;
    const int kbA     = laneHi * 8;    // A-fragment K sub-offset (ISA 16-bit A layout)
    const int kbB     = laneHi * 8;    // B-fragment K sub-offset
    const int hidBase = wave * 32;
    const int m0      = blockIdx.x * ROWS;
    const int nt      = wave & 3;      // logits N-tile
    const int kq      = wave >> 2;     // logits K-quarter

    // ---- setup ----
    for (int j = tid; j < ROWS * HID; j += THREADS)
        ((unsigned short*)h_bf_s)[j] = 0;
    if (tid < ROWS) tok_s[tid] = x[m0 + tid];
    if (tid < OUTW) blin_s[tid] = b_lin[tid];

    // per-lane gate biases (8 registers, resident for the whole kernel)
    float bias_r[8];
    #pragma unroll
    for (int g = 0; g < 4; ++g)
        #pragma unroll
        for (int t = 0; t < 2; ++t) {
            int n = g * HID + hidBase + t * 16 + ln;
            bias_r[g * 2 + t] = b_ih[n] + b_hh[n];
        }

    v8f creg[2];                       // cell state, registers for all 20 steps
    #pragma unroll
    for (int t = 0; t < 2; ++t)
        #pragma unroll
        for (int e = 0; e < 8; ++e) creg[t][e] = 0.0f;

    __syncthreads();

    for (int step = 0; step < STEPS; ++step) {
        // Opaque zero per iteration: blocks LICM of the (step-invariant) weight
        // loads out of the step loop, which previously caused scratch spills.
        int opq;
        asm volatile("s_mov_b32 %0, 0" : "=s"(opq));
        const unsigned short* wih_p  = wih_bf  + opq;
        const unsigned short* whh_p  = whh_bf  + opq;
        const unsigned short* wlin_p = wlin_bf + opq;
        const unsigned short* emb_p  = emb_bf  + opq;

        // ---- gates GEMM: acc[g*2+t] covers columns g*512 + hidBase + t*16 + ln ----
        v8f acc[8];
        #pragma unroll
        for (int i = 0; i < 8; ++i)
            #pragma unroll
            for (int e = 0; e < 8; ++e) acc[i][e] = 0.0f;

        // phase 1: embedding rows (K = 0..127), A from global (L2-resident)
        {
            const unsigned short* arow = emb_p + (size_t)tok_s[ln] * EMB;
            #pragma unroll
            for (int kk = 0; kk < EMB; kk += 32) {
                v16bf af = load_frag16(arow + kk + kbA, arow + kk + kbA + 16);
                #pragma unroll
                for (int g = 0; g < 4; ++g)
                    #pragma unroll
                    for (int t = 0; t < 2; ++t) {
                        int n = g * HID + hidBase + t * 16 + ln;
                        const unsigned short* brow = wih_p + (size_t)n * EMB + kk + kbB;
                        v16bf bfrag = load_frag16(brow, brow + 16);
                        acc[g * 2 + t] = __builtin_amdgcn_wmma_f32_16x16x32_bf16(
                            false, af, false, bfrag, (short)0, acc[g * 2 + t], false, false);
                    }
            }
        }
        // phase 2: recurrent h (K = 0..511), A from LDS (ds_read_b128)
        {
            const unsigned short* arow = &h_bf_s[ln][0];
            #pragma unroll 4
            for (int kk = 0; kk < HID; kk += 32) {
                v16bf af = load_frag16(arow + kk + kbA, arow + kk + kbA + 16);
                #pragma unroll
                for (int g = 0; g < 4; ++g)
                    #pragma unroll
                    for (int t = 0; t < 2; ++t) {
                        int n = g * HID + hidBase + t * 16 + ln;
                        const unsigned short* brow = whh_p + (size_t)n * HID + kk + kbB;
                        v16bf bfrag = load_frag16(brow, brow + 16);
                        acc[g * 2 + t] = __builtin_amdgcn_wmma_f32_16x16x32_bf16(
                            false, af, false, bfrag, (short)0, acc[g * 2 + t], false, false);
                    }
            }
        }
        __syncthreads();   // all waves finished reading old h_bf_s / tok_s

        // ---- LSTM elementwise (torch gate order i,f,g,o), write new h (bf16) ----
        #pragma unroll
        for (int t = 0; t < 2; ++t) {
            #pragma unroll
            for (int e = 0; e < 8; ++e) {
                float iv = acc[0 + t][e] + bias_r[0 + t];
                float fv = acc[2 + t][e] + bias_r[2 + t];
                float gv = acc[4 + t][e] + bias_r[4 + t];
                float ov = acc[6 + t][e] + bias_r[6 + t];
                float cn = sigf(fv) * creg[t][e] + sigf(iv) * tanhfast(gv);
                float hv = sigf(ov) * tanhfast(cn);
                creg[t][e] = cn;
                int hr = e + laneHi * 8;           // C-layout: VGPR e -> M=e (+8 hi half)
                int hc = hidBase + t * 16 + ln;
                h_bf_s[hr][hc] = f2bf(hv);
            }
        }
        __syncthreads();

        // ---- logits via WMMA: [16 x 64] = h[16 x 512] @ W_lin^T ----
        // wave w: N-tile nt = w&3 (cols nt*16..), K-quarter kq = w>>2 (K kq*128..)
        {
            v8f lacc;
            #pragma unroll
            for (int e = 0; e < 8; ++e) lacc[e] = 0.0f;
            const unsigned short* arow = &h_bf_s[ln][0];
            const int n = nt * 16 + ln;
            #pragma unroll
            for (int kk = 0; kk < 128; kk += 32) {
                int k = kq * 128 + kk;
                v16bf af = load_frag16(arow + k + kbA, arow + k + kbA + 16);
                const unsigned short* brow = wlin_p + (size_t)n * HID + k + kbB;
                v16bf bfrag = load_frag16(brow, brow + 16);
                lacc = __builtin_amdgcn_wmma_f32_16x16x32_bf16(
                    false, af, false, bfrag, (short)0, lacc, false, false);
            }
            #pragma unroll
            for (int e = 0; e < 8; ++e)
                partial_s[kq][e + laneHi * 8][nt * 16 + ln] = lacc[e];
        }
        __syncthreads();

        // ---- reduce K-quarters + bias -> logits ----
        #pragma unroll
        for (int p = 0; p < 2; ++p) {
            int idx = tid + p * THREADS;
            int row = idx >> 6;
            int o   = idx & (OUTW - 1);
            logits_s[row][o] = blin_s[o]
                             + partial_s[0][row][o] + partial_s[1][row][o]
                             + partial_s[2][row][o] + partial_s[3][row][o];
        }
        __syncthreads();

        // ---- softmax stats + deterministic categorical sample (1 thread/row) ----
        if (tid < ROWS) {
            float mx = -3.4e38f;
            for (int o = 0; o < OUTW; ++o) mx = fmaxf(mx, logits_s[tid][o]);
            float sum = 0.0f;
            for (int o = 0; o < OUTW; ++o) sum += __expf(logits_s[tid][o] - mx);
            float inv = 1.0f / sum;
            unsigned int h = (unsigned int)((m0 + tid) * 32 + step + 1);
            h *= 2654435761u; h ^= h >> 16; h *= 0x85ebca6bu;
            h ^= h >> 13;     h *= 0xc2b2ae35u; h ^= h >> 16;
            float u = (float)(h >> 8) * (1.0f / 16777216.0f);
            float cum = 0.0f;
            int sel = OUTW - 1;
            for (int o = 0; o < OUTW; ++o) {
                cum += __expf(logits_s[tid][o] - mx) * inv;
                if (u < cum) { sel = o; break; }
            }
            tok_s[tid] = sel;
            rmax_s[tid] = mx;
            rinv_s[tid] = inv;
        }
        __syncthreads();

        // ---- write probs[b, step, :] ----
        #pragma unroll
        for (int p = 0; p < 2; ++p) {
            int idx = tid + p * THREADS;
            int row = idx >> 6;
            int o   = idx & (OUTW - 1);
            float pr = __expf(logits_s[row][o] - rmax_s[row]) * rinv_s[row];
            out[((size_t)(m0 + row) * STEPS + step) * OUTW + o] = pr;
        }
    }
}

extern "C" void kernel_launch(void* const* d_in, const int* in_sizes, int n_in,
                              void* d_out, int out_size, void* d_ws, size_t ws_size,
                              hipStream_t stream) {
    (void)in_sizes; (void)n_in; (void)out_size; (void)ws_size;
    const int*   x    = (const int*)d_in[0];
    const float* emb  = (const float*)d_in[1];
    const float* Wih  = (const float*)d_in[2];
    const float* Whh  = (const float*)d_in[3];
    const float* bih  = (const float*)d_in[4];
    const float* bhh  = (const float*)d_in[5];
    const float* Wlin = (const float*)d_in[6];
    const float* blin = (const float*)d_in[7];
    float* out = (float*)d_out;

    unsigned short* wih_bf  = (unsigned short*)d_ws;        // 2048*128
    unsigned short* whh_bf  = wih_bf + GATES * EMB;         // 2048*512
    unsigned short* emb_bf  = whh_bf + GATES * HID;         // 64*128
    unsigned short* wlin_bf = emb_bf + OUTW * EMB;          // 64*512

    int total = GATES * EMB + GATES * HID + OUTW * EMB + OUTW * HID;
    convert_bf16_kernel<<<(total + 255) / 256, 256, 0, stream>>>(
        Wih, Whh, emb, Wlin, wih_bf, whh_bf, emb_bf, wlin_bf);

    controller_kernel<<<B_TOT / ROWS, THREADS, 0, stream>>>(
        x, emb_bf, wih_bf, whh_bf, wlin_bf, bih, bhh, blin, out);
}
